// MessagePackBlock_76046690943129
// MI455X (gfx1250) — compile-verified, matrix operand order checked by reading, not run on hardware
//
#include <hip/hip_runtime.h>
#include <hip/hip_bf16.h>
#include <cstdint>

typedef __attribute__((ext_vector_type(16))) _Float16 v16h;
typedef __attribute__((ext_vector_type(8)))  float    v8f;

#define TILE_E   16
#define NTHREADS 256

#define RSQRT3_F  0.5773502691896258f
#define RSQRT8_F  0.3535533905932738f
#define RSQRT32_F 0.17677669529663687f

__device__ __forceinline__ float silu_f(float x) {
    return x / (1.0f + __expf(-x));
}

struct Params {
    const float* src;   // node_feats_src  (E,128)
    const float* dst;   // node_feats_dst  (E,128)
    const float* ef;    // edge_feats      (E,128)
    const float* le;    // local_env_edge  (E,4)
    const float* sc;    // edge_scalars    (E,8)
    const float* Wn1; const float* Wn2; const float* Wn3; const float* Wn4; // (64,32)
    const float* We1; const float* We2; const float* We3; const float* We4; // (32,32)
    const float* Mn1; const float* Mn2;  // (8,64), (64,64)
    const float* Me1; const float* Me2;
    const float* Lns; const float* Lnv; const float* Les; const float* Lev; // (32,32)
    const _Float16* wf; // pre-packed f16 B fragments of Wn_mlp3 / We_mlp3
    float* out;         // (E,128)
};

// A/B 16-bit WMMA fragment K pattern (CDNA5 ISA 16-bit 16x32 layout):
// lane l, half i holds K = (l<16 ? 0 : 8) + (i<8 ? i : i+8)
__device__ __forceinline__ int kpat(int lane, int i) {
    return ((lane < 16) ? 0 : 8) + ((i < 8) ? i : i + 8);
}

// ---------------------------------------------------------------------------
// Prep: Wn_mlp3 / We_mlp3 (f32, 64x4096) -> f16 B fragments in register
// order (scale 1/8 folded in).  Fragment idx = (p*256 + nTile)*2 + kStep,
// 32 lanes x 16 halves each (32 contiguous bytes per lane).  1 MB in d_ws.
// ---------------------------------------------------------------------------
__global__ __launch_bounds__(NTHREADS)
void prep_w3_kernel(const float* __restrict__ Wn3,
                    const float* __restrict__ We3,
                    _Float16* __restrict__ wf) {
    int gid = blockIdx.x * NTHREADS + threadIdx.x;   // 0 .. 32767
    if (gid >= 2 * 256 * 2 * 32) return;
    int lane = gid & 31;
    int ks   = (gid >> 5) & 1;
    int tile = (gid >> 6) & 255;
    int p    = gid >> 14;
    const float* W = p ? We3 : Wn3;
    int n = tile * 16 + (lane & 15);
    _Float16* dstp = wf + (size_t)gid * 16;
#pragma unroll
    for (int i = 0; i < 16; ++i) {
        int K = ks * 32 + kpat(lane, i);
        dstp[i] = (_Float16)(W[K * 4096 + n] * 0.125f);
    }
}

// ---------------------------------------------------------------------------
// Fused kernel: one workgroup (256 thr = 8 waves) handles TILE_E=16 edges.
// ---------------------------------------------------------------------------
__global__ __launch_bounds__(NTHREADS)
void fused_message_kernel(Params P, int E) {
    const int t    = threadIdx.x;
    const int wave = t >> 5;
    const int lane = t & 31;
    const int e0   = blockIdx.x * TILE_E;

    __shared__ float s1b[TILE_E][64];
    __shared__ float v1b[TILE_E][64][3];
    __shared__ float msb[TILE_E][64];
    __shared__ float mvb[TILE_E][64][3];
    __shared__ float h1b[TILE_E][64];
    __shared__ float h2b[TILE_E][64];
    __shared__ float scb[TILE_E][8];
    __shared__ float lsb[TILE_E];
    __shared__ float lvb[TILE_E][3];
    __shared__ float accS[2][TILE_E][32];
    __shared__ float accV[2][TILE_E][32][3];

    // ---- Phase 0 ----
    if (t < TILE_E * 8) {
        int r = t >> 3, i = t & 7;
        int er = min(e0 + r, E - 1);
        scb[r][i] = P.sc[er * 8 + i];
    }
    if (t < TILE_E * 4) {
        int r = t >> 2, m = t & 3;
        int er = min(e0 + r, E - 1);
        float v = P.le[er * 4 + m];
        if (m == 0) lsb[r] = v; else lvb[r][m - 1] = v;
    }
    for (int idx = t; idx < 2 * TILE_E * 32; idx += NTHREADS)
        ((float*)accS)[idx] = 0.0f;
    for (int idx = t; idx < 2 * TILE_E * 32 * 3; idx += NTHREADS)
        ((float*)accV)[idx] = 0.0f;
    __syncthreads();

    for (int p = 0; p < 2; ++p) {
        const int   Up    = (p == 0) ? 64 : 32;
        const float alpha = (p == 0) ? 0.125f : RSQRT32_F;
        const float* Wt1 = p ? P.We1 : P.Wn1;
        const float* Wt2 = p ? P.We2 : P.Wn2;
        const float* Wt3 = p ? P.We3 : P.Wn3;
        const float* Wt4 = p ? P.We4 : P.Wn4;
        const float* M1  = p ? P.Me1 : P.Mn1;
        const float* M2  = p ? P.Me2 : P.Mn2;

        // ---- Phase A: stage s1/v1 into LDS ----
        if (p == 0) {
            for (int idx = t; idx < TILE_E * 128; idx += NTHREADS) {
                int r = idx >> 7, c = idx & 127;
                int er = min(e0 + r, E - 1);
                float a = P.src[er * 128 + c];
                float b = P.dst[er * 128 + c];
                if (c < 32) { s1b[r][c] = a; s1b[r][32 + c] = b; }
                else {
                    int q = c - 32, u = q / 3, m = q - 3 * u;
                    v1b[r][u][m] = a; v1b[r][32 + u][m] = b;
                }
            }
        } else {
            for (int idx = t; idx < TILE_E * 128; idx += NTHREADS) {
                int r = idx >> 7, c = idx & 127;
                int er = min(e0 + r, E - 1);
                float a = P.ef[er * 128 + c];
                if (c < 32) s1b[r][c] = a;
                else {
                    int q = c - 32, u = q / 3, m = q - 3 * u;
                    v1b[r][u][m] = a;
                }
            }
        }
        __syncthreads();

        // ---- Phase B: tensor product -> ms (16x64), mv (16x64x3) ----
#pragma unroll
        for (int rr = 0; rr < 2; ++rr) {
            int r = wave + rr * 8;
            int w = lane;
            float l0 = lvb[r][0], l1 = lvb[r][1], l2 = lvb[r][2];
            float lsr = lsb[r];
            float a0 = 0.f, a2 = 0.f, b0 = 0.f, b1 = 0.f, b2 = 0.f, d = 0.f;
            for (int u = 0; u < Up; ++u) {
                float s  = s1b[r][u];
                float x0 = v1b[r][u][0], x1 = v1b[r][u][1], x2 = v1b[r][u][2];
                a0 += s * Wt1[u * 32 + w];
                a2 += s * Wt2[u * 32 + w];
                float w3 = Wt3[u * 32 + w];
                b0 += x0 * w3; b1 += x1 * w3; b2 += x2 * w3;
                float du = (x0 * l0 + x1 * l1 + x2 * l2) * RSQRT3_F;
                d += du * Wt4[u * 32 + w];
            }
            msb[r][w]         = a0 * lsr * alpha;
            msb[r][32 + w]    = d * alpha;
            mvb[r][w][0]      = a2 * l0 * alpha;
            mvb[r][w][1]      = a2 * l1 * alpha;
            mvb[r][w][2]      = a2 * l2 * alpha;
            mvb[r][32 + w][0] = b0 * lsr * alpha;
            mvb[r][32 + w][1] = b1 * lsr * alpha;
            mvb[r][32 + w][2] = b2 * lsr * alpha;
        }

        // ---- Phase C1: h1 = silu(sc @ M1/sqrt(8)) ----
        {
            int j = t & 63;
            for (int r = t >> 6; r < TILE_E; r += 4) {
                float a = 0.f;
#pragma unroll
                for (int i = 0; i < 8; ++i) a += scb[r][i] * M1[i * 64 + j];
                h1b[r][j] = silu_f(a * RSQRT8_F);
            }
        }
        __syncthreads();

        // ---- Phase C2: h2 = silu(h1 @ M2/8) ----
        {
            int j = t & 63;
            for (int r = t >> 6; r < TILE_E; r += 4) {
                float a = 0.f;
                for (int k = 0; k < 64; ++k) a += h1b[r][k] * M2[k * 64 + j];
                h2b[r][j] = silu_f(a * 0.125f);
            }
        }
        __syncthreads();

        // ---- Phase D: A fragments (h2, 16x64 f16) ----
        v16h af0, af1;
        {
            int row = lane & 15;
#pragma unroll
            for (int i = 0; i < 16; ++i) {
                int kk = kpat(lane, i);
                af0[i] = (_Float16)h2b[row][kk];
                af1[i] = (_Float16)h2b[row][32 + kk];
            }
        }

        // ---- Phase E: 16x64 @ 64x4096 GEMM fused with lin_scale ----
        // waves 0..3 -> scalar half (cols 0..2047); 4..7 -> vector half.
        const bool isvec = (wave >= 4);
        const int  ebase = (lane < 16) ? 0 : 8;
        const int  col   = lane & 15;
        const int  t0    = wave * 32;
        const _Float16* wfp = P.wf + (size_t)p * (256 * 2 * 32 * 16);

        // fragment pointer: tile index, k-step
#define FRAGP(tile_, ks_) \
        ((const v16h*)(wfp + (((size_t)((tile_) * 2 + (ks_))) * 32 + lane) * 16))

        if (!isvec) {
            float ps0[8], ps1[8];
#pragma unroll
            for (int r = 0; r < 8; ++r) { ps0[r] = 0.f; ps1[r] = 0.f; }
            float msr[8];
            v16h c0 = *FRAGP(t0, 0);
            v16h c1 = *FRAGP(t0, 1);
#pragma unroll 2
            for (int ti = 0; ti < 32; ti += 2) {
                int tile = t0 + ti;
                int u = tile >> 1;              // constant across the pair
#pragma unroll
                for (int r = 0; r < 8; ++r) msr[r] = msb[ebase + r][u];
                // fragments of the odd tile while even tile computes
                v16h d0 = *FRAGP(tile + 1, 0);
                v16h d1 = *FRAGP(tile + 1, 1);
                __builtin_prefetch((const void*)FRAGP(tile + 2, 0), 0, 1);
                v8f acc = {};
                acc = __builtin_amdgcn_wmma_f32_16x16x32_f16(
                    false, af0, false, c0, (short)0, acc, false, false);
                acc = __builtin_amdgcn_wmma_f32_16x16x32_f16(
                    false, af1, false, c1, (short)0, acc, false, false);
#pragma unroll
                for (int r = 0; r < 8; ++r) ps0[r] += acc[r] * msr[r];
                if (ti + 2 < 32) { c0 = *FRAGP(tile + 2, 0); c1 = *FRAGP(tile + 2, 1); }
                v8f acc2 = {};
                acc2 = __builtin_amdgcn_wmma_f32_16x16x32_f16(
                    false, af0, false, d0, (short)0, acc2, false, false);
                acc2 = __builtin_amdgcn_wmma_f32_16x16x32_f16(
                    false, af1, false, d1, (short)0, acc2, false, false);
#pragma unroll
                for (int r = 0; r < 8; ++r) ps1[r] += acc2[r] * msr[r];
            }
#pragma unroll
            for (int r = 0; r < 8; ++r) {
                atomicAdd(&accS[p][ebase + r][col],      ps0[r]);
                atomicAdd(&accS[p][ebase + r][16 + col], ps1[r]);
            }
        } else {
            float pv0[8][3], pv1[8][3];
#pragma unroll
            for (int r = 0; r < 8; ++r)
#pragma unroll
                for (int m = 0; m < 3; ++m) { pv0[r][m] = 0.f; pv1[r][m] = 0.f; }
            float mvr[8][3];
            v16h c0 = *FRAGP(t0, 0);
            v16h c1 = *FRAGP(t0, 1);
#pragma unroll 2
            for (int ti = 0; ti < 32; ti += 2) {
                int tile = t0 + ti;
                int u = (tile - 128) >> 1;
#pragma unroll
                for (int r = 0; r < 8; ++r) {
                    mvr[r][0] = mvb[ebase + r][u][0];
                    mvr[r][1] = mvb[ebase + r][u][1];
                    mvr[r][2] = mvb[ebase + r][u][2];
                }
                v16h d0 = *FRAGP(tile + 1, 0);
                v16h d1 = *FRAGP(tile + 1, 1);
                __builtin_prefetch((const void*)FRAGP(tile + 2, 0), 0, 1);
                v8f acc = {};
                acc = __builtin_amdgcn_wmma_f32_16x16x32_f16(
                    false, af0, false, c0, (short)0, acc, false, false);
                acc = __builtin_amdgcn_wmma_f32_16x16x32_f16(
                    false, af1, false, c1, (short)0, acc, false, false);
#pragma unroll
                for (int r = 0; r < 8; ++r) {
                    pv0[r][0] += acc[r] * mvr[r][0];
                    pv0[r][1] += acc[r] * mvr[r][1];
                    pv0[r][2] += acc[r] * mvr[r][2];
                }
                if (ti + 2 < 32) { c0 = *FRAGP(tile + 2, 0); c1 = *FRAGP(tile + 2, 1); }
                v8f acc2 = {};
                acc2 = __builtin_amdgcn_wmma_f32_16x16x32_f16(
                    false, af0, false, d0, (short)0, acc2, false, false);
                acc2 = __builtin_amdgcn_wmma_f32_16x16x32_f16(
                    false, af1, false, d1, (short)0, acc2, false, false);
#pragma unroll
                for (int r = 0; r < 8; ++r) {
                    pv1[r][0] += acc2[r] * mvr[r][0];
                    pv1[r][1] += acc2[r] * mvr[r][1];
                    pv1[r][2] += acc2[r] * mvr[r][2];
                }
            }
#pragma unroll
            for (int r = 0; r < 8; ++r)
#pragma unroll
                for (int m = 0; m < 3; ++m) {
                    atomicAdd(&accV[p][ebase + r][col][m],      pv0[r][m]);
                    atomicAdd(&accV[p][ebase + r][16 + col][m], pv1[r][m]);
                }
        }
#undef FRAGP
        __syncthreads();
    }

    // ---- Phase F: lin_out (32x32) for both paths, sum, store (E,128) ----
    for (int idx = t; idx < TILE_E * 32; idx += NTHREADS) {
        int e = idx >> 5, w = idx & 31;
        if (e0 + e >= E) continue;
        float ss = 0.f, vx = 0.f, vy = 0.f, vz = 0.f;
        for (int k = 0; k < 32; ++k) {
            float lns = P.Lns[k * 32 + w], les = P.Les[k * 32 + w];
            ss += accS[0][e][k] * lns + accS[1][e][k] * les;
            float lnv = P.Lnv[k * 32 + w], lev = P.Lev[k * 32 + w];
            vx += accV[0][e][k][0] * lnv + accV[1][e][k][0] * lev;
            vy += accV[0][e][k][1] * lnv + accV[1][e][k][1] * lev;
            vz += accV[0][e][k][2] * lnv + accV[1][e][k][2] * lev;
        }
        float* o = P.out + (size_t)(e0 + e) * 128;
        o[w]              = ss * RSQRT32_F;
        o[32 + w * 3 + 0] = vx * RSQRT32_F;
        o[32 + w * 3 + 1] = vy * RSQRT32_F;
        o[32 + w * 3 + 2] = vz * RSQRT32_F;
    }
}

// ---------------------------------------------------------------------------
extern "C" void kernel_launch(void* const* d_in, const int* in_sizes, int n_in,
                              void* d_out, int out_size, void* d_ws, size_t ws_size,
                              hipStream_t stream) {
    const int E = in_sizes[3] / 4;   // local_env_edge is (E,4)

    Params P;
    P.src = (const float*)d_in[0];
    P.dst = (const float*)d_in[1];
    P.ef  = (const float*)d_in[2];
    P.le  = (const float*)d_in[3];
    P.sc  = (const float*)d_in[4];
    P.Wn1 = (const float*)d_in[5];
    P.Wn2 = (const float*)d_in[6];
    P.Wn3 = (const float*)d_in[7];
    P.Wn4 = (const float*)d_in[8];
    P.We1 = (const float*)d_in[9];
    P.We2 = (const float*)d_in[10];
    P.We3 = (const float*)d_in[11];
    P.We4 = (const float*)d_in[12];
    P.Mn1 = (const float*)d_in[13];
    P.Mn2 = (const float*)d_in[14];
    const float* Wn_mlp3 = (const float*)d_in[15];
    P.Me1 = (const float*)d_in[16];
    P.Me2 = (const float*)d_in[17];
    const float* We_mlp3 = (const float*)d_in[18];
    P.Lns = (const float*)d_in[19];
    P.Lnv = (const float*)d_in[20];
    P.Les = (const float*)d_in[21];
    P.Lev = (const float*)d_in[22];
    P.wf  = (const _Float16*)d_ws;   // needs 1 MB of scratch
    P.out = (float*)d_out;

    prep_w3_kernel<<<128, NTHREADS, 0, stream>>>(Wn_mlp3, We_mlp3,
                                                 (_Float16*)d_ws);
    const int blocks = (E + TILE_E - 1) / TILE_E;   // 1875 for E=30000
    fused_message_kernel<<<blocks, NTHREADS, 0, stream>>>(P, E);
}
